// EmformerBlock_28733331210928
// MI455X (gfx1250) — matrix operand order repaired and checked by reading, not compiled, not run on hardware
//
#include <hip/hip_runtime.h>

// ---------------------------------------------------------------------------
// MI455X (gfx1250) transformer block: LN -> QKV -> flash attention -> LN ->
// FFN -> LN.  GEMMs: v_wmma_f32_16x16x32_bf16 (fp32 accumulate).
// Global->LDS staging uses CDNA5 async-to-LDS DMA (ASYNCcnt), double-buffered
// so the next tile's DMA overlaps the current tile's WMMAs.
// Weights are transposed+cast once so B fragments stream from L2 directly.
// wave32, 256-thread blocks (8 waves).
// ---------------------------------------------------------------------------

typedef __attribute__((ext_vector_type(16))) __bf16 v16bf;
typedef __attribute__((ext_vector_type(8)))  float  v8f;

union Frag { v16bf v; uint4 q[2]; };

#define BDIM  512      // D
#define TDIM  1024     // T
#define FDIM  2048     // F
#define HNUM  8
#define DKDIM 64
#define MROWS 8192     // B*T
#define LNEPS 1e-3f
#define NEGBIG (-1.0e9f)

// ---------------------------------------------------------------------------
// CDNA5 async global->LDS copy (16 bytes per lane), tracked by ASYNCcnt.
// Generic pointers to __shared__ carry the LDS byte offset in their low 32
// bits, so the truncated pointer is the VDST LDS address the instruction
// wants.  Completion is in-order per wave, so waiting asynccnt<=N with N
// newer copies in flight guarantees the older tile has landed.
// ---------------------------------------------------------------------------
__device__ __forceinline__ void async_copy16(void* lds, const void* gptr) {
    unsigned l = (unsigned)(unsigned long long)lds;
    asm volatile("global_load_async_to_lds_b128 %0, %1, off"
                 :: "v"(l), "v"(gptr) : "memory");
}
__device__ __forceinline__ void async_wait0() {
    asm volatile("s_wait_asynccnt 0x0" ::: "memory");
}
__device__ __forceinline__ void async_wait1() {
    asm volatile("s_wait_asynccnt 0x1" ::: "memory");
}
__device__ __forceinline__ void async_wait2() {
    asm volatile("s_wait_asynccnt 0x2" ::: "memory");
}

// ---------------------------------------------------------------------------
// fp32 -> bf16 transpose+cast: Wt[n][k] = (bf16) W[k][n].  32x32 LDS tile.
// ---------------------------------------------------------------------------
__global__ __launch_bounds__(256)
void transpose_cast_kernel(const float* __restrict__ W, __bf16* __restrict__ Wt,
                           int K, int N) {
    __shared__ float tile[32][33];
    const int kb = blockIdx.y * 32, nb = blockIdx.x * 32;
    const int tx = threadIdx.x & 31, ty = threadIdx.x >> 5;   // 32 x 8
    #pragma unroll
    for (int i = 0; i < 4; ++i)
        tile[ty + i * 8][tx] = W[(size_t)(kb + ty + i * 8) * N + nb + tx];
    __syncthreads();
    #pragma unroll
    for (int i = 0; i < 4; ++i)
        Wt[(size_t)(nb + ty + i * 8) * K + kb + tx] = (__bf16)tile[tx][ty + i * 8];
}

// ---------------------------------------------------------------------------
// LayerNorm over rows of length 512. One 256-thread block per row.
// ---------------------------------------------------------------------------
__global__ __launch_bounds__(256)
void layernorm_kernel(const float* __restrict__ in, const float* __restrict__ g,
                      const float* __restrict__ bvec, float* __restrict__ out_f,
                      __bf16* __restrict__ out_b) {
    const int row = blockIdx.x;
    const float* r = in + (size_t)row * BDIM;
    const int tid = threadIdx.x;
    float v0 = r[tid], v1 = r[tid + 256];

    __shared__ float red[2][8];
    float s = v0 + v1, sq = v0 * v0 + v1 * v1;
    #pragma unroll
    for (int off = 16; off >= 1; off >>= 1) {
        s  += __shfl_xor(s,  off, 32);
        sq += __shfl_xor(sq, off, 32);
    }
    const int wave = tid >> 5, lane = tid & 31;
    if (lane == 0) { red[0][wave] = s; red[1][wave] = sq; }
    __syncthreads();
    if (wave == 0) {
        float s2 = red[0][lane & 7], q2 = red[1][lane & 7];
        #pragma unroll
        for (int off = 4; off >= 1; off >>= 1) {
            s2 += __shfl_xor(s2, off, 32);
            q2 += __shfl_xor(q2, off, 32);
        }
        if (lane == 0) { red[0][0] = s2; red[1][0] = q2; }
    }
    __syncthreads();
    const float mean = red[0][0] * (1.0f / BDIM);
    const float var  = red[1][0] * (1.0f / BDIM) - mean * mean;
    const float rstd = rsqrtf(var + LNEPS);

    #pragma unroll
    for (int i = 0; i < 2; ++i) {
        const int c = tid + i * 256;
        const float val = (r[c] - mean) * rstd * g[c] + bvec[c];
        if (out_f) out_f[(size_t)row * BDIM + c] = val;
        if (out_b) out_b[(size_t)row * BDIM + c] = (__bf16)val;
    }
}

// ---------------------------------------------------------------------------
// WMMA GEMM: C[M,N] = A[M,K] @ W[K,N] + bias.  A bf16 row-major, weights
// pre-transposed Wt[N,K] bf16 so B fragments are two aligned b128 global
// loads per lane (hot in the 192MB L2).  A tiles are async-DMA'd into a
// double-buffered LDS pipeline.  Block tile 64x64, K-step 32, 8 waves.
// MODE: 1 = bf16 out, head layout [B,H,T,DK]     (Q, K)
//       2 = bf16 out, transposed head [B,H,DK,T] (V)
//       3 = bf16 out, relu, row-major            (FFN1)
//       4 = fp32 out + residual add, row-major   (FFN2)
// ---------------------------------------------------------------------------
template <int MODE>
__global__ __launch_bounds__(256)
void gemm_bf16(const __bf16* __restrict__ A, const __bf16* __restrict__ Wt,
               const float* __restrict__ bias, const float* __restrict__ resid,
               void* __restrict__ Cout, int N, int K) {
    __shared__ __align__(16) __bf16 As[2][64 * 40];   // [m][k], padded stride 40

    const int tid  = threadIdx.x;
    const int lane = tid & 31;
    const int wave = tid >> 5;
    const int mBase = blockIdx.y * 64;
    const int nBase = blockIdx.x * 64;
    const int wm = wave & 3;            // m-subtile row (16 rows)
    const int wn = (wave >> 2) * 2;     // first of two n-subtiles

    const int laneLo = lane & 15;
    const int laneHi = lane >> 4;
    const int klo = laneHi * 8;         // A-frag element offset
    const int kbo = laneHi * 16;        // unused; B frags also use klo

    // each wave's two B columns (constant pointers, advance by k0)
    const __bf16* bcol0 = Wt + (size_t)(nBase + wn * 16 + laneLo) * K;
    const __bf16* bcol1 = Wt + (size_t)(nBase + (wn + 1) * 16 + laneLo) * K;
    // per-thread A staging assignment: one b128 per thread per K-step
    const int ar = tid >> 2;            // m row 0..63
    const int ac = (tid & 3) * 8;       // k col 0,8,16,24
    const __bf16* Abase = A + (size_t)(mBase + ar) * K + ac;
    const int ldsOff = ar * 40 + ac;

    v8f acc0 = {}, acc1 = {};
    const int ktot = K >> 5;

    async_copy16(&As[0][ldsOff], Abase);            // prologue: tile 0
    for (int kt = 0; kt < ktot; ++kt) {
        const int cur = kt & 1;
        const int k0 = kt * 32;
        if (kt + 1 < ktot) {                        // prefetch tile kt+1
            async_copy16(&As[cur ^ 1][ldsOff], Abase + (kt + 1) * 32);
            async_wait1();                          // tile kt landed
        } else {
            async_wait0();
        }
        __syncthreads();

        Frag fa, fb0, fb1;
        const __bf16* ap = &As[cur][(wm * 16 + laneLo) * 40];
        fa.q[0] = *reinterpret_cast<const uint4*>(ap + klo);
        fa.q[1] = *reinterpret_cast<const uint4*>(ap + 16 + klo);
        fb0.q[0] = *reinterpret_cast<const uint4*>(bcol0 + k0 + klo);
        fb0.q[1] = *reinterpret_cast<const uint4*>(bcol0 + k0 + 16 + klo);
        fb1.q[0] = *reinterpret_cast<const uint4*>(bcol1 + k0 + klo);
        fb1.q[1] = *reinterpret_cast<const uint4*>(bcol1 + k0 + 16 + klo);

        acc0 = __builtin_amdgcn_wmma_f32_16x16x32_bf16(false, fa.v, false, fb0.v,
                                                       (short)0, acc0, false, false);
        acc1 = __builtin_amdgcn_wmma_f32_16x16x32_bf16(false, fa.v, false, fb1.v,
                                                       (short)0, acc1, false, false);
        __syncthreads();   // all reads of As[cur] done before it is re-filled
    }

    // epilogue: C layout element (e, lane) -> M = e + 8*laneHi, N = laneLo
    #pragma unroll
    for (int sub = 0; sub < 2; ++sub) {
        const v8f& acc = sub ? acc1 : acc0;
        const int nt = wn + sub;
        #pragma unroll
        for (int e = 0; e < 8; ++e) {
            const int m = mBase + wm * 16 + e + 8 * laneHi;
            const int n = nBase + nt * 16 + laneLo;
            float v = acc[e] + bias[n];
            if constexpr (MODE == 1) {
                const int bb = m >> 10, t = m & 1023, hh = n >> 6, dk = n & 63;
                ((__bf16*)Cout)[((((size_t)bb * HNUM + hh) << 10) + t) * DKDIM + dk] = (__bf16)v;
            } else if constexpr (MODE == 2) {
                const int bb = m >> 10, t = m & 1023, hh = n >> 6, dk = n & 63;
                ((__bf16*)Cout)[((((size_t)bb * HNUM + hh) * DKDIM + dk) << 10) + t] = (__bf16)v;
            } else if constexpr (MODE == 3) {
                ((__bf16*)Cout)[(size_t)m * N + n] = (__bf16)fmaxf(v, 0.0f);
            } else {
                ((float*)Cout)[(size_t)m * N + n] = v + resid[(size_t)m * N + n];
            }
        }
    }
}

// ---------------------------------------------------------------------------
// Flash attention per (b,h).  Block = 8 waves, each owns 16 query rows
// (128 rows/block); all waves share the same K/V tiles, async-DMA'd into a
// double-buffered LDS pipeline (2 copies/wave/tile -> wait asynccnt<=2).
// Online softmax fp32; P@V via WMMA with an LDS bounce converting C-layout
// probs to A-layout bf16.  Output: attnres = attn + x  (fp32, [B,T,D]).
// ---------------------------------------------------------------------------
__global__ __launch_bounds__(256)
void attention_kernel(const __bf16* __restrict__ qbuf, const __bf16* __restrict__ kbuf,
                      const __bf16* __restrict__ vbuf, const float* __restrict__ mask,
                      const float* __restrict__ x, float* __restrict__ attnres) {
    __shared__ __align__(16) __bf16 Ks[2][32 * 64];     // [key][d]
    __shared__ __align__(16) __bf16 Vs[2][64 * 32];     // [d][key]  (from V^T)
    __shared__ __align__(16) __bf16 Pst[8][16 * 40];    // per-wave P tile [16][32]

    const int tid  = threadIdx.x;
    const int lane = tid & 31;
    const int wave = tid >> 5;
    const int bh = blockIdx.y;
    const int b = bh >> 3, h = bh & 7;
    const int qBase = blockIdx.x * 128 + wave * 16;
    const int laneLo = lane & 15;
    const int laneHi = lane >> 4;
    const int klo = laneHi * 8;      // A-frag offset
    const int kbo = laneHi * 16;     // unused; B frags also use klo

    // Q fragments (A-matrix), d = 0..31 and 32..63
    Frag aq0, aq1;
    {
        const __bf16* qp = qbuf + ((size_t)bh * TDIM + qBase + laneLo) * DKDIM;
        aq0.q[0] = *(const uint4*)(qp + klo);
        aq0.q[1] = *(const uint4*)(qp + 16 + klo);
        aq1.q[0] = *(const uint4*)(qp + 32 + klo);
        aq1.q[1] = *(const uint4*)(qp + 48 + klo);
    }

    // per-thread K/V staging assignment (one b128 each per tile)
    const int kr = tid >> 3, kc = (tid & 7) * 8;    // K tile: 32 keys x 64 d
    const int vr = tid >> 2, vc = (tid & 3) * 8;    // V tile: 64 d x 32 keys
    const __bf16* kgbase = kbuf + ((size_t)bh * TDIM + kr) * DKDIM + kc;
    const __bf16* vgbase = vbuf + ((size_t)bh * DKDIM + vr) * TDIM + vc;
    const int kOff = kr * 64 + kc, vOff = vr * 32 + vc;

    v8f acc[4] = {};                 // O accumulator: 4 d-chunks of 16
    float mrun[8], lrun[8];
    #pragma unroll
    for (int e = 0; e < 8; ++e) { mrun[e] = -3.0e38f; lrun[e] = 0.0f; }

    const float scale = 0.125f;      // 1/sqrt(64)
    const int stot = TDIM / 32;

    async_copy16(&Ks[0][kOff], kgbase);             // prologue: tile 0
    async_copy16(&Vs[0][vOff], vgbase);
    for (int st = 0; st < stot; ++st) {
        const int cur = st & 1;
        const int s0 = st * 32;
        if (st + 1 < stot) {                        // prefetch tile st+1
            async_copy16(&Ks[cur ^ 1][kOff], kgbase + (size_t)(s0 + 32) * DKDIM);
            async_copy16(&Vs[cur ^ 1][vOff], vgbase + s0 + 32);
            async_wait2();                          // tile st landed
        } else {
            async_wait0();
        }
        __syncthreads();

        float p[2][8];
        #pragma unroll
        for (int half = 0; half < 2; ++half) {
            const int key = s0 + half * 16 + laneLo;
            const __bf16* kp = &Ks[cur][(half * 16 + laneLo) * 64];
            Frag bk0, bk1;                          // B-matrix = K^T columns
            bk0.q[0] = *(const uint4*)(kp + klo);
            bk0.q[1] = *(const uint4*)(kp + 16 + klo);
            bk1.q[0] = *(const uint4*)(kp + 32 + klo);
            bk1.q[1] = *(const uint4*)(kp + 48 + klo);
            v8f S = {};
            S = __builtin_amdgcn_wmma_f32_16x16x32_bf16(false, aq0.v, false, bk0.v,
                                                        (short)0, S, false, false);
            S = __builtin_amdgcn_wmma_f32_16x16x32_bf16(false, aq1.v, false, bk1.v,
                                                        (short)0, S, false, false);
            #pragma unroll
            for (int e = 0; e < 8; ++e) {
                const int row = qBase + e + 8 * laneHi;
                const float mv = mask[((size_t)b * TDIM + row) * TDIM + key];
                p[half][e] = S[e] * scale + NEGBIG * (1.0f - mv);
            }
        }
        // online softmax statistics (rows split across 16-lane halves)
        float mnew[8], resc[8];
        #pragma unroll
        for (int e = 0; e < 8; ++e) {
            float mx = fmaxf(p[0][e], p[1][e]);
            #pragma unroll
            for (int off = 1; off < 16; off <<= 1) mx = fmaxf(mx, __shfl_xor(mx, off, 32));
            mnew[e] = fmaxf(mrun[e], mx);
            resc[e] = __expf(mrun[e] - mnew[e]);
            mrun[e] = mnew[e];
        }
        #pragma unroll
        for (int e = 0; e < 8; ++e) {
            p[0][e] = __expf(p[0][e] - mnew[e]);
            p[1][e] = __expf(p[1][e] - mnew[e]);
            float sum = p[0][e] + p[1][e];
            #pragma unroll
            for (int off = 1; off < 16; off <<= 1) sum += __shfl_xor(sum, off, 32);
            lrun[e] = lrun[e] * resc[e] + sum;
        }
        #pragma unroll
        for (int c = 0; c < 4; ++c)
            #pragma unroll
            for (int e = 0; e < 8; ++e) acc[c][e] *= resc[e];

        // bounce P (C-layout fp32) through LDS into A-layout bf16
        __bf16* ps = &Pst[wave][0];
        #pragma unroll
        for (int e = 0; e < 8; ++e) {
            const int row = e + 8 * laneHi;
            ps[row * 40 + laneLo]      = (__bf16)p[0][e];
            ps[row * 40 + 16 + laneLo] = (__bf16)p[1][e];
        }
        Frag ap;
        const __bf16* pr = &Pst[wave][laneLo * 40];
        ap.q[0] = *(const uint4*)(pr + klo);
        ap.q[1] = *(const uint4*)(pr + 16 + klo);

        // O += P @ V   (Vs rows: d, contiguous over keys)
        #pragma unroll
        for (int c = 0; c < 4; ++c) {
            const __bf16* vp = &Vs[cur][(c * 16 + laneLo) * 32];
            Frag fv;
            fv.q[0] = *(const uint4*)(vp + klo);
            fv.q[1] = *(const uint4*)(vp + 16 + klo);
            acc[c] = __builtin_amdgcn_wmma_f32_16x16x32_bf16(false, ap.v, false, fv.v,
                                                             (short)0, acc[c], false, false);
        }
        __syncthreads();   // all reads of Ks/Vs[cur] done before refill
    }

    // normalize, add residual x, write [B,T,D]
    float inv[8];
    #pragma unroll
    for (int e = 0; e < 8; ++e) inv[e] = 1.0f / lrun[e];
    #pragma unroll
    for (int c = 0; c < 4; ++c) {
        #pragma unroll
        for (int e = 0; e < 8; ++e) {
            const int row = qBase + e + 8 * laneHi;
            const int d = h * DKDIM + c * 16 + laneLo;
            const size_t idx = ((size_t)b * TDIM + row) * BDIM + d;
            attnres[idx] = acc[c][e] * inv[e] + x[idx];
        }
    }
}

// ---------------------------------------------------------------------------
// Host-side orchestration
// ---------------------------------------------------------------------------
extern "C" void kernel_launch(void* const* d_in, const int* in_sizes, int n_in,
                              void* d_out, int out_size, void* d_ws, size_t ws_size,
                              hipStream_t stream) {
    const float* x       = (const float*)d_in[0];
    const float* mask    = (const float*)d_in[1];
    const float* ln_in_g = (const float*)d_in[2];
    const float* ln_in_b = (const float*)d_in[3];
    const float* Wq      = (const float*)d_in[4];
    const float* bq      = (const float*)d_in[5];
    const float* Wk      = (const float*)d_in[6];
    const float* bk      = (const float*)d_in[7];
    const float* Wv      = (const float*)d_in[8];
    const float* bv      = (const float*)d_in[9];
    const float* ln1_g   = (const float*)d_in[10];
    const float* ln1_b   = (const float*)d_in[11];
    const float* W1      = (const float*)d_in[12];
    const float* b1      = (const float*)d_in[13];
    const float* W2      = (const float*)d_in[14];
    const float* b2      = (const float*)d_in[15];
    const float* ln2_g   = (const float*)d_in[16];
    const float* ln2_b   = (const float*)d_in[17];
    float* out = (float*)d_out;

    char* ws = (char*)d_ws;
    const size_t SZ_HB = (size_t)MROWS * BDIM * 2;        // 8 MiB bf16 [M,D]
    // region0 (4 x SZ_HB = 32 MiB): h,q,k,v during attention; f1 during FFN
    __bf16* hb  = (__bf16*)(ws + 0 * SZ_HB);
    __bf16* qb  = (__bf16*)(ws + 1 * SZ_HB);
    __bf16* kb  = (__bf16*)(ws + 2 * SZ_HB);
    __bf16* vtb = (__bf16*)(ws + 3 * SZ_HB);
    __bf16* f1b = (__bf16*)(ws + 0);                      // reuse region0
    char* p = ws + 4 * SZ_HB;
    float* attnres = (float*)p;  p += (size_t)MROWS * BDIM * 4;
    float* y = attnres;                                   // reuse
    float* a = (float*)p;        p += (size_t)MROWS * BDIM * 4;
    __bf16* ab = (__bf16*)p;     p += SZ_HB;
    __bf16* Wqt = (__bf16*)p;    p += (size_t)BDIM * BDIM * 2;   // [N,K] bf16
    __bf16* Wkt = (__bf16*)p;    p += (size_t)BDIM * BDIM * 2;
    __bf16* Wvt = (__bf16*)p;    p += (size_t)BDIM * BDIM * 2;
    __bf16* W1t = (__bf16*)p;    p += (size_t)BDIM * FDIM * 2;   // [F,D]
    __bf16* W2t = (__bf16*)p;    p += (size_t)FDIM * BDIM * 2;   // [D,F]

    // 1. weight transpose + cast fp32 -> bf16  (Wt[n][k] = W[k][n])
    transpose_cast_kernel<<<dim3(BDIM / 32, BDIM / 32), 256, 0, stream>>>(Wq, Wqt, BDIM, BDIM);
    transpose_cast_kernel<<<dim3(BDIM / 32, BDIM / 32), 256, 0, stream>>>(Wk, Wkt, BDIM, BDIM);
    transpose_cast_kernel<<<dim3(BDIM / 32, BDIM / 32), 256, 0, stream>>>(Wv, Wvt, BDIM, BDIM);
    transpose_cast_kernel<<<dim3(FDIM / 32, BDIM / 32), 256, 0, stream>>>(W1, W1t, BDIM, FDIM);
    transpose_cast_kernel<<<dim3(BDIM / 32, FDIM / 32), 256, 0, stream>>>(W2, W2t, FDIM, BDIM);

    // 2. input LayerNorm -> bf16 h
    layernorm_kernel<<<MROWS, 256, 0, stream>>>(x, ln_in_g, ln_in_b, nullptr, hb);

    // 3. QKV projections (WMMA)
    dim3 gQKV(BDIM / 64, MROWS / 64);
    gemm_bf16<1><<<gQKV, 256, 0, stream>>>(hb, Wqt, bq, nullptr, qb,  BDIM, BDIM);
    gemm_bf16<1><<<gQKV, 256, 0, stream>>>(hb, Wkt, bk, nullptr, kb,  BDIM, BDIM);
    gemm_bf16<2><<<gQKV, 256, 0, stream>>>(hb, Wvt, bv, nullptr, vtb, BDIM, BDIM);

    // 4. flash attention + residual(x)
    attention_kernel<<<dim3(TDIM / 128, 8 * HNUM), 256, 0, stream>>>(qb, kb, vtb,
                                                                     mask, x, attnres);

    // 5. LN1 -> a (fp32) and ab (bf16)
    layernorm_kernel<<<MROWS, 256, 0, stream>>>(attnres, ln1_g, ln1_b, a, ab);

    // 6. FFN1: relu(a @ W1 + b1) -> bf16
    gemm_bf16<3><<<dim3(FDIM / 64, MROWS / 64), 256, 0, stream>>>(ab, W1t, b1,
                                                                  nullptr, f1b, FDIM, BDIM);
    // 7. FFN2: f1 @ W2 + b2 + a -> fp32 y
    gemm_bf16<4><<<dim3(BDIM / 64, MROWS / 64), 256, 0, stream>>>(f1b, W2t, b2,
                                                                  a, y, BDIM, FDIM);
    // 8. LN2 -> out
    layernorm_kernel<<<MROWS, 256, 0, stream>>>(y, ln2_g, ln2_b, out, nullptr);
}
